// BidirectionalCrossAttention_59940563583465
// MI455X (gfx1250) — compile-verified, matrix operand order checked by reading, not run on hardware
//
#include <hip/hip_runtime.h>

typedef __attribute__((ext_vector_type(16))) __bf16 v16bf;
typedef __attribute__((ext_vector_type(8)))  float  v8f;

#define NSEQ   1024
#define DMODEL 1024

struct alignas(8) US4 { unsigned short x, y, z, w; };

__device__ __forceinline__ unsigned short f2bf(float f) {
  unsigned int u = __float_as_uint(f);
  u += 0x7FFFu + ((u >> 16) & 1u);          // round-to-nearest-even
  return (unsigned short)(u >> 16);
}
__device__ __forceinline__ float bf2f(unsigned short h) {
  return __uint_as_float(((unsigned int)h) << 16);
}

// Load one 16x32 bf16 WMMA A/B fragment (row-major, leading dim ld).
// ISA layout (16-bit 16x32): lanes 0-15 -> K {0..7,16..23}; lanes 16-31 -> K {8..15,24..31}.
__device__ __forceinline__ v16bf load_frag(const unsigned short* __restrict__ base,
                                           int ld, int rowBase, int kk, int lane) {
  int sel = lane >> 4;
  int r   = lane & 15;
  const unsigned short* p = base + (size_t)(rowBase + r) * ld + kk + sel * 8;
  union { uint4 q[2]; v16bf v; } u;
  u.q[0] = *(const uint4*)(p);
  u.q[1] = *(const uint4*)(p + 16);
  return u.v;
}

// Fragment read from LDS (ld = 32 elements, kk = 0 within staged tile).
__device__ __forceinline__ v16bf lds_frag(const unsigned short* p) {
  union { uint4 q[2]; v16bf v; } u;
  u.q[0] = *(const uint4*)(p);
  u.q[1] = *(const uint4*)(p + 16);
  return u.v;
}

// Async global -> LDS 16-byte copy (CDNA5, ASYNCcnt-tracked).
__device__ __forceinline__ void async_ld_b128(unsigned lds_off, const unsigned short* g) {
  asm volatile("global_load_async_to_lds_b128 %0, %1, off"
               :: "v"(lds_off), "v"(g) : "memory");
}
__device__ __forceinline__ void wait_async0() {
  asm volatile("s_wait_asynccnt 0x0" ::: "memory");
}

#define WMMA_BF16(d, a, b) \
  d = __builtin_amdgcn_wmma_f32_16x16x32_bf16(false, a, false, b, (short)0, d, false, false)

// ---------------------------------------------------------------------------
// Direct-from-global WMMA GEMM (used for sim and attention-out GEMMs).
// C[M x N] = A[M x K] * Bt[N x K]^T, fp32 accumulate; wave tile 32x64.
// mode 0: bf16 -> obf[row*ldc + colOff+col]
// mode 2: fp32 val*scale -> of32[row*ldc + col]
// ---------------------------------------------------------------------------
__global__ __launch_bounds__(256) void wmma_gemm_kernel(
    const unsigned short* __restrict__ A, int lda,
    const unsigned short* __restrict__ Bt, int ldb,
    int tilesM, int tilesN, int K,
    int mode, float scale,
    unsigned short* __restrict__ obf, int ldc, int colOff,
    float* __restrict__ of32)
{
  int wid = blockIdx.x * 8 + (threadIdx.x >> 5);
  if (wid >= tilesM * tilesN) return;              // whole-wave uniform exit
  int tm = wid / tilesN, tn = wid - tm * tilesN;
  int lane = threadIdx.x & 31;
  int rowA = tm * 32, rowB = tn * 64;

  v8f z = {};
  v8f acc[2][4];
#pragma unroll
  for (int i = 0; i < 2; ++i)
#pragma unroll
    for (int j = 0; j < 4; ++j) acc[i][j] = z;

  for (int kk = 0; kk < K; kk += 32) {
    v16bf a0 = load_frag(A, lda, rowA,      kk, lane);
    v16bf a1 = load_frag(A, lda, rowA + 16, kk, lane);
    v16bf b0 = load_frag(Bt, ldb, rowB,      kk, lane);
    v16bf b1 = load_frag(Bt, ldb, rowB + 16, kk, lane);
    v16bf b2 = load_frag(Bt, ldb, rowB + 32, kk, lane);
    v16bf b3 = load_frag(Bt, ldb, rowB + 48, kk, lane);
    WMMA_BF16(acc[0][0], a0, b0); WMMA_BF16(acc[0][1], a0, b1);
    WMMA_BF16(acc[0][2], a0, b2); WMMA_BF16(acc[0][3], a0, b3);
    WMMA_BF16(acc[1][0], a1, b0); WMMA_BF16(acc[1][1], a1, b1);
    WMMA_BF16(acc[1][2], a1, b2); WMMA_BF16(acc[1][3], a1, b3);
  }

  int sel = lane >> 4, r = lane & 15;
#pragma unroll
  for (int mi = 0; mi < 2; ++mi) {
#pragma unroll
    for (int q = 0; q < 4; ++q) {
      int col  = rowB + q * 16 + r;
      int gcol = colOff + col;
#pragma unroll
      for (int e = 0; e < 8; ++e) {
        int row = rowA + mi * 16 + sel * 8 + e;
        float val = acc[mi][q][e];
        if (mode == 0) obf[(size_t)row * ldc + gcol] = f2bf(val);
        else           of32[(size_t)row * ldc + col] = val * scale;
      }
    }
  }
}

// ---------------------------------------------------------------------------
// Block-cooperative WMMA GEMM with async-to-LDS double-buffered staging.
// Fixed: lda = ldb = ldc = 1024, block tile 64(M) x 256(N), K multiple of 32.
// 8 waves (2x4), wave tile 32x64. LDS: 2 x (64x32 A + 256x32 B) bf16 = 40 KB.
// mode 0: bf16 -> obf[row*1024 + col]
// mode 1: mode 0 + transposed copy aux[(row>>10)<<20 | col<<10 | row&1023]
// mode 3: fp32 val + bias[col] -> of32[row*1024 + col]
// ---------------------------------------------------------------------------
#define TILE_ELEMS (320 * 32)   // (64 + 256) rows x 32 cols per buffer
__global__ __launch_bounds__(256) void wmma_gemm_lds_kernel(
    const unsigned short* __restrict__ A,
    const unsigned short* __restrict__ Bt,
    int nBn, int K, int mode,
    unsigned short* __restrict__ obf,
    float* __restrict__ of32, const float* __restrict__ bias,
    unsigned short* __restrict__ aux)
{
  __shared__ unsigned short sh[2][TILE_ELEMS];   // [buf][A(64x32) | B(256x32)]

  int bm = blockIdx.x / nBn, bn = blockIdx.x - bm * nBn;
  int rowA0 = bm * 64, rowB0 = bn * 256;
  int t = threadIdx.x;

  // Loader mapping: 5 x 16B async copies per thread per k-step.
  int lr = t >> 2;          // 0..63
  int lc = (t & 3) * 8;     // element offset of 16B chunk within 32-col row
  const unsigned short* gA = A  + (size_t)(rowA0 + lr) * DMODEL + lc;
  const unsigned short* gB = Bt + (size_t)(rowB0 + lr) * DMODEL + lc;

  auto issue = [&](int buf, int kk) {
    async_ld_b128((unsigned)(uintptr_t)&sh[buf][lr * 32 + lc], gA + kk);
#pragma unroll
    for (int q = 0; q < 4; ++q)
      async_ld_b128((unsigned)(uintptr_t)&sh[buf][2048 + (lr + q * 64) * 32 + lc],
                    gB + (size_t)q * 64 * DMODEL + kk);
  };

  // Compute mapping.
  int w    = t >> 5;        // wave 0..7
  int wm   = w >> 2;        // 0..1
  int wn   = w & 3;         // 0..3
  int lane = t & 31;
  int sel  = lane >> 4, r = lane & 15;

  v8f z = {};
  v8f acc[2][4];
#pragma unroll
  for (int i = 0; i < 2; ++i)
#pragma unroll
    for (int j = 0; j < 4; ++j) acc[i][j] = z;

  issue(0, 0);
  int buf = 0;
  for (int kk = 0; kk < K; kk += 32) {
    wait_async0();            // this wave's fills of `buf` have landed
    __syncthreads();          // everyone's fills landed; everyone done reading buf^1
    if (kk + 32 < K) issue(buf ^ 1, kk + 32);

    const unsigned short* sA = &sh[buf][0];
    const unsigned short* sB = &sh[buf][2048];
    v16bf a0 = lds_frag(sA + (wm * 32 +      r) * 32 + sel * 8);
    v16bf a1 = lds_frag(sA + (wm * 32 + 16 + r) * 32 + sel * 8);
    v16bf b0 = lds_frag(sB + (wn * 64 +      r) * 32 + sel * 8);
    v16bf b1 = lds_frag(sB + (wn * 64 + 16 + r) * 32 + sel * 8);
    v16bf b2 = lds_frag(sB + (wn * 64 + 32 + r) * 32 + sel * 8);
    v16bf b3 = lds_frag(sB + (wn * 64 + 48 + r) * 32 + sel * 8);
    WMMA_BF16(acc[0][0], a0, b0); WMMA_BF16(acc[0][1], a0, b1);
    WMMA_BF16(acc[0][2], a0, b2); WMMA_BF16(acc[0][3], a0, b3);
    WMMA_BF16(acc[1][0], a1, b0); WMMA_BF16(acc[1][1], a1, b1);
    WMMA_BF16(acc[1][2], a1, b2); WMMA_BF16(acc[1][3], a1, b3);
    buf ^= 1;
  }

  int rBase = rowA0 + wm * 32;
  int cBase = rowB0 + wn * 64;
#pragma unroll
  for (int mi = 0; mi < 2; ++mi) {
#pragma unroll
    for (int q = 0; q < 4; ++q) {
      int col = cBase + q * 16 + r;
#pragma unroll
      for (int e = 0; e < 8; ++e) {
        int row = rBase + mi * 16 + sel * 8 + e;
        float val = acc[mi][q][e];
        if (mode <= 1) {
          obf[(size_t)row * DMODEL + col] = f2bf(val);
          if (mode == 1)
            aux[(((size_t)(row >> 10)) << 20) + (((size_t)col) << 10) + (row & 1023)] = f2bf(val);
        } else {
          of32[(size_t)row * DMODEL + col] = val + bias[col];
        }
      }
    }
  }
}

// LayerNorm over 1024 cols, one block per row, bf16 output.
__global__ __launch_bounds__(256) void ln_bf16_kernel(
    const float* __restrict__ X, const float* __restrict__ G,
    const float* __restrict__ Bv, unsigned short* __restrict__ out)
{
  int row = blockIdx.x, tid = threadIdx.x;
  const float4 x = *(const float4*)(X + (size_t)row * DMODEL + tid * 4);
  __shared__ float s1[256], s2[256];
  s1[tid] = x.x + x.y + x.z + x.w;
  s2[tid] = x.x * x.x + x.y * x.y + x.z * x.z + x.w * x.w;
  __syncthreads();
  for (int o = 128; o > 0; o >>= 1) {
    if (tid < o) { s1[tid] += s1[tid + o]; s2[tid] += s2[tid + o]; }
    __syncthreads();
  }
  float mu   = s1[0] * (1.0f / DMODEL);
  float var  = s2[0] * (1.0f / DMODEL) - mu * mu;
  float rstd = rsqrtf(var + 1e-5f);
  const float4 g  = *(const float4*)(G + tid * 4);
  const float4 bb = *(const float4*)(Bv + tid * 4);
  US4 o4;
  o4.x = f2bf((x.x - mu) * rstd * g.x + bb.x);
  o4.y = f2bf((x.y - mu) * rstd * g.y + bb.y);
  o4.z = f2bf((x.z - mu) * rstd * g.z + bb.z);
  o4.w = f2bf((x.w - mu) * rstd * g.w + bb.w);
  *(US4*)(out + (size_t)row * DMODEL + tid * 4) = o4;
}

// Wt[n*K + k] = bf16(W[k*N + n])
__global__ __launch_bounds__(256) void wtrans_kernel(
    const float* __restrict__ W, unsigned short* __restrict__ Wt, int K, int N)
{
  size_t idx = (size_t)blockIdx.x * 256 + threadIdx.x;
  int n = (int)(idx / K);
  int k = (int)(idx % K);
  Wt[idx] = f2bf(W[(size_t)k * N + n]);
}

// Row softmax (unnormalized): P = bf16(exp(x - rowmax)), rs[row] = row sum.
__global__ __launch_bounds__(256) void softmax_kernel(
    const float* __restrict__ sim, unsigned short* __restrict__ P,
    float* __restrict__ rs)
{
  int i = blockIdx.x, tid = threadIdx.x;
  const float4 x = *(const float4*)(sim + (size_t)i * NSEQ + tid * 4);
  __shared__ float red[256];
  red[tid] = fmaxf(fmaxf(x.x, x.y), fmaxf(x.z, x.w));
  __syncthreads();
  for (int o = 128; o > 0; o >>= 1) {
    if (tid < o) red[tid] = fmaxf(red[tid], red[tid + o]);
    __syncthreads();
  }
  float M = red[0];
  __syncthreads();
  float e0 = __expf(x.x - M), e1 = __expf(x.y - M);
  float e2 = __expf(x.z - M), e3 = __expf(x.w - M);
  red[tid] = e0 + e1 + e2 + e3;
  __syncthreads();
  for (int o = 128; o > 0; o >>= 1) {
    if (tid < o) red[tid] += red[tid + o];
    __syncthreads();
  }
  if (tid == 0) rs[i] = red[0];
  US4 o4; o4.x = f2bf(e0); o4.y = f2bf(e1); o4.z = f2bf(e2); o4.w = f2bf(e3);
  *(US4*)(P + (size_t)i * NSEQ + tid * 4) = o4;
}

// Talking-heads mix: Mx[h][i][j] = bf16( sum_g (Wth[h][g]/rs[g][i]) * P[g][i][j] )
__global__ __launch_bounds__(256) void mix_kernel(
    const unsigned short* __restrict__ P, const float* __restrict__ rs,
    const float* __restrict__ Wth, unsigned short* __restrict__ Mx)
{
  int i = blockIdx.x, tid = threadIdx.x;
  __shared__ float coeff[256];
  {
    int h = tid >> 4, g = tid & 15;
    coeff[tid] = Wth[h * 16 + g] / rs[(size_t)g * NSEQ + i];
  }
  __syncthreads();
#pragma unroll
  for (int jj = 0; jj < 4; ++jj) {
    int j = tid + jj * 256;
    float pv[16];
#pragma unroll
    for (int g = 0; g < 16; ++g)
      pv[g] = bf2f(P[(((size_t)g) << 20) + (size_t)i * NSEQ + j]);
#pragma unroll
    for (int h = 0; h < 16; ++h) {
      float a = 0.f;
#pragma unroll
      for (int g = 0; g < 16; ++g) a = fmaf(coeff[h * 16 + g], pv[g], a);
      Mx[(((size_t)h) << 20) + (size_t)i * NSEQ + j] = f2bf(a);
    }
  }
}

extern "C" void kernel_launch(void* const* d_in, const int* in_sizes, int n_in,
                              void* d_out, int out_size, void* d_ws, size_t ws_size,
                              hipStream_t stream)
{
  (void)in_sizes; (void)n_in; (void)out_size; (void)ws_size;
  const float* text    = (const float*)d_in[0];
  const float* audio   = (const float*)d_in[1];
  const float* g_text  = (const float*)d_in[2];
  const float* b_text  = (const float*)d_in[3];
  const float* g_audio = (const float*)d_in[4];
  const float* b_audio = (const float*)d_in[5];
  const float* W_qk    = (const float*)d_in[6];
  const float* W_v     = (const float*)d_in[7];
  const float* W_out   = (const float*)d_in[8];
  const float* b_out   = (const float*)d_in[9];
  const float* W_th    = (const float*)d_in[10];
  float* out = (float*)d_out;

  // Workspace layout (~122 MB)
  unsigned short* XT   = (unsigned short*)d_ws;            // text LN bf16   (4096x1024)
  unsigned short* XA   = XT   + (size_t)4096 * 1024;       // audio LN bf16
  unsigned short* WQKT = XA   + (size_t)4096 * 1024;       // W_qk^T bf16 (1024x1024)
  unsigned short* WVT  = WQKT + (size_t)1024 * 1024;       // W_v^T  bf16
  unsigned short* WOT  = WVT  + (size_t)1024 * 1024;       // W_out^T bf16
  unsigned short* QK   = WOT  + (size_t)1024 * 1024;       // qk bf16 (b,i,h*64+d)
  unsigned short* V    = QK   + (size_t)4096 * 1024;       // v  bf16 (b,j,h*64+d)
  unsigned short* VT   = V    + (size_t)4096 * 1024;       // v^T bf16 (b, h*64+d, j)
  unsigned short* AOUT = VT   + (size_t)4096 * 1024;       // attn out bf16 (b,i,h*64+d)
  float*          SIM  = (float*)(AOUT + (size_t)4096 * 1024);          // per-(b,h) sim fp32
  unsigned short* P    = (unsigned short*)(SIM + (size_t)1024 * 1024);  // exp probs bf16 (16,1024,1024)
  float*          RS   = (float*)(P + (size_t)16 * 1024 * 1024);        // row sums (16,1024)
  unsigned short* MX   = (unsigned short*)(RS + 16 * 1024);             // mixed attn bf16 (16,1024,1024)

  dim3 blk(256);

  ln_bf16_kernel<<<4096, blk, 0, stream>>>(text,  g_text,  b_text,  XT);
  ln_bf16_kernel<<<4096, blk, 0, stream>>>(audio, g_audio, b_audio, XA);

  wtrans_kernel<<<4096, blk, 0, stream>>>(W_qk,  WQKT, 1024, 1024);
  wtrans_kernel<<<4096, blk, 0, stream>>>(W_v,   WVT,  1024, 1024);
  wtrans_kernel<<<4096, blk, 0, stream>>>(W_out, WOT,  1024, 1024);

  // qk = audio_ln @ W_qk   (M=4096, N=1024, K=1024): 64x4 block tiles
  wmma_gemm_lds_kernel<<<256, blk, 0, stream>>>(XA, WQKT, 4, 1024,
      0, QK, nullptr, nullptr, nullptr);
  // v = text_ln @ W_v  (+ transposed copy VT)
  wmma_gemm_lds_kernel<<<256, blk, 0, stream>>>(XT, WVT, 4, 1024,
      1, V, nullptr, nullptr, VT);

  const float scale = 0.125f;  // DIM_HEAD^-0.5
  for (int b = 0; b < 4; ++b) {
    for (int h = 0; h < 16; ++h) {
      // sim = qk[b,h] @ v[b,h]^T * scale  (M=N=1024, K=64)
      wmma_gemm_kernel<<<64, blk, 0, stream>>>(
          QK + (size_t)b * 1048576 + h * 64, 1024,
          V  + (size_t)b * 1048576 + h * 64, 1024,
          32, 16, 64, 2, scale, nullptr, 1024, 0, SIM);
      softmax_kernel<<<1024, blk, 0, stream>>>(SIM, P + (((size_t)h) << 20), RS + h * 1024);
    }
    mix_kernel<<<1024, blk, 0, stream>>>(P, RS, W_th, MX);
    // out[b,h] = mixed[h] @ v[b,h]  (M=1024, N=64, K=1024), B^T = VT slice
    for (int h = 0; h < 16; ++h) {
      wmma_gemm_kernel<<<4, blk, 0, stream>>>(
          MX + (((size_t)h) << 20), 1024,
          VT + ((size_t)b * 16 + h) * 65536, 1024,
          32, 1, 1024, 0, 1.f,
          AOUT + (size_t)b * 1048576, 1024, h * 64, nullptr);
    }
  }

  // final = attn_out @ W_out + b_out  (fp32 out)
  wmma_gemm_lds_kernel<<<256, blk, 0, stream>>>(AOUT, WOT, 4, 1024,
      3, nullptr, out, b_out, nullptr);
}
